// LSTMDecoder_10711648436649
// MI455X (gfx1250) — matrix operand order repaired and checked by reading, not compiled, not run on hardware
//
#include <hip/hip_runtime.h>

// ---------------------------------------------------------------------------
// LSTM decoder for MI455X (gfx1250): bf16 WMMA (f32 accum) for both the
// hoisted input projection and the sequential recurrence.
//   L=128, B=64, IN_F=ACT_F=512, LSTM_IN=1024, H=1024, 4H=4096
// Wave = 2 M-tiles x 4 N-tiles (8 acc chains). Rolled K loop, ping-pong
// double buffer; ALL stream addresses are immediate offsets from just two
// base pointers (fits global_load's 24-bit imm) -> no spills, minimal VALU.
// ---------------------------------------------------------------------------

#define L_STEPS 128
#define BATCH   64
#define HID     1024
#define G4H     4096          // 4*H
#define KDIM    1024          // LSTM_IN == H == 1024 (both GEMMs share K)
#define MTOT    (L_STEPS * BATCH)   // 8192 rows of X
#define AROW1   (16 * KDIM)   // element offset of 2nd A row-tile (32768 B)

typedef __bf16 v8bf  __attribute__((ext_vector_type(8)));
typedef __bf16 v16bf __attribute__((ext_vector_type(16)));
typedef float  v8f   __attribute__((ext_vector_type(8)));

// ---- workspace layout (bytes) ---------------------------------------------
#define OFF_WIH  (size_t)0                       // 4096*1024 bf16 =   8 MB
#define OFF_WHH  (size_t)(8u  * 1024 * 1024)     // 4096*1024 bf16 =   8 MB
#define OFF_X    (size_t)(16u * 1024 * 1024)     // 8192*1024 bf16 =  16 MB
#define OFF_GIN  (size_t)(32u * 1024 * 1024)     // 8192*4096 f32  = 128 MB
#define OFF_BIAS (OFF_GIN + (size_t)MTOT * G4H * 4)          // 4096 f32
#define OFF_H    (OFF_BIAS + (size_t)G4H * 4)                // 64*1024 bf16
#define OFF_C    (OFF_H + (size_t)BATCH * HID * 2)           // 64*1024 f32

__device__ __forceinline__ float fsigmoid(float x) {
    return 1.0f / (1.0f + __expf(-x));
}
__device__ __forceinline__ float ftanh(float x) {
    float e = __expf(2.0f * x);
    return (e - 1.0f) / (e + 1.0f);
}

// A fragment (16-bit A-matrix 16x32 layout, ISA 7.12.2), from pointer already
// offset for this lane: elements 0..7 at p, 8..15 at p+16 (bf16 elements).
__device__ __forceinline__ v16bf load_a_p(const __bf16* p) {
    v8bf lo = *(const v8bf*)(p);
    v8bf hi = *(const v8bf*)(p + 16);
    return __builtin_shufflevector(lo, hi, 0,1,2,3,4,5,6,7,8,9,10,11,12,13,14,15);
}
// B fragment (column n of B = row n of W): elements 0..7 at p, 8..15 at p+16.
__device__ __forceinline__ v16bf load_b_p(const __bf16* p) {
    v8bf lo = *(const v8bf*)(p);
    v8bf hi = *(const v8bf*)(p + 16);
    return __builtin_shufflevector(lo, hi, 0,1,2,3,4,5,6,7,8,9,10,11,12,13,14,15);
}

#define WMMA_BF16(A, Bm, C) \
    __builtin_amdgcn_wmma_f32_16x16x32_bf16(false, (A), false, (Bm), (short)0, (C), false, false)

#define WMMA8(A0_, A1_, B0_, B1_, B2_, B3_)      \
    do {                                         \
        acc0 = WMMA_BF16(A0_, B0_, acc0);        \
        acc1 = WMMA_BF16(A0_, B1_, acc1);        \
        acc2 = WMMA_BF16(A0_, B2_, acc2);        \
        acc3 = WMMA_BF16(A0_, B3_, acc3);        \
        acc4 = WMMA_BF16(A1_, B0_, acc4);        \
        acc5 = WMMA_BF16(A1_, B1_, acc5);        \
        acc6 = WMMA_BF16(A1_, B2_, acc6);        \
        acc7 = WMMA_BF16(A1_, B3_, acc7);        \
    } while (0)

// ---------------------------------------------------------------------------
// Shared GEMM core: 8 accumulators over K=1024, ping-pong double buffer.
// pa: A base (lane-offset); second A row-tile at +AROW1 elements (32768 B imm).
// pb: B base (lane-offset); streams at +0,+BS,+2*BS,+3*BS elements (24-bit imm
//     for both BS=16*KDIM [gemm, 32768 B] and BS=HID*KDIM [step, 2 MB]).
// Only pa/pb are incremented (+64 elements/iter); every load is base+imm.
// NOTE: final even-chunk loads are dead prefetches 64B past the tile; all such
// reads stay inside the workspace (X->Gin, Wih->Whh, Whh->X, hB->cS).
// ---------------------------------------------------------------------------
#define GEMM_K_LOOP(pa, pb, BS)                                         \
    {                                                                   \
        v16bf a0 = load_a_p(pa);                                        \
        v16bf a1 = load_a_p(pa + AROW1);                                \
        v16bf b0 = load_b_p(pb);                                        \
        v16bf b1 = load_b_p(pb + (BS));                                 \
        v16bf b2 = load_b_p(pb + 2 * (BS));                             \
        v16bf b3 = load_b_p(pb + 3 * (BS));                             \
        _Pragma("unroll 1")                                             \
        for (int it = 0; it < KDIM / 64; ++it) {                        \
            /* odd chunk (+32 elements) */                              \
            v16bf oa0 = load_a_p(pa + 32);                              \
            v16bf oa1 = load_a_p(pa + AROW1 + 32);                      \
            v16bf ob0 = load_b_p(pb + 32);                              \
            v16bf ob1 = load_b_p(pb + (BS) + 32);                       \
            v16bf ob2 = load_b_p(pb + 2 * (BS) + 32);                   \
            v16bf ob3 = load_b_p(pb + 3 * (BS) + 32);                   \
            WMMA8(a0, a1, b0, b1, b2, b3);                              \
            /* next even chunk (+64 elements); dead prefetch at end */  \
            pa += 64;                                                   \
            pb += 64;                                                   \
            a0 = load_a_p(pa);                                          \
            a1 = load_a_p(pa + AROW1);                                  \
            b0 = load_b_p(pb);                                          \
            b1 = load_b_p(pb + (BS));                                   \
            b2 = load_b_p(pb + 2 * (BS));                               \
            b3 = load_b_p(pb + 3 * (BS));                               \
            WMMA8(oa0, oa1, ob0, ob1, ob2, ob3);                        \
        }                                                               \
    }

// ---------------------------------------------------------------------------
// Prep kernels
// ---------------------------------------------------------------------------
__global__ void k_prep_weights(const float* __restrict__ Wih,
                               const float* __restrict__ Whh,
                               const float* __restrict__ bih,
                               const float* __restrict__ bhh,
                               __bf16* __restrict__ WihB,
                               __bf16* __restrict__ WhhB,
                               float* __restrict__ bias) {
    size_t i = (size_t)blockIdx.x * blockDim.x + threadIdx.x;   // 4096*1024 threads
    WihB[i] = (__bf16)Wih[i];
    WhhB[i] = (__bf16)Whh[i];
    if (i < G4H) bias[i] = bih[i] + bhh[i];
}

__global__ void k_prep_x(const float* __restrict__ in_feat,   // [64][512]
                         const float* __restrict__ act,       // [128][64][512]
                         __bf16* __restrict__ X) {            // [8192][1024]
    size_t i = (size_t)blockIdx.x * blockDim.x + threadIdx.x;  // 8192*1024 threads
    int k = (int)(i & (KDIM - 1));
    size_t m = i >> 10;                 // t*64 + b
    int b = (int)(m & (BATCH - 1));
    size_t t = m >> 6;
    float v = (k < 512) ? in_feat[(size_t)b * 512 + k]
                        : act[(t * BATCH + b) * 512 + (k - 512)];
    X[i] = (__bf16)v;
}

__global__ void k_init_state(const float* __restrict__ h0,
                             const float* __restrict__ c0,
                             __bf16* __restrict__ hB,
                             float* __restrict__ cS) {
    size_t i = (size_t)blockIdx.x * blockDim.x + threadIdx.x;  // 64*1024
    hB[i] = (__bf16)h0[i];
    cS[i] = c0[i];
}

// ---------------------------------------------------------------------------
// Big input-projection GEMM: Gin[m][n] = X[m][:] . Wih[n][:] + bias[n]
// Wave -> 32-row x 64-col output block. 256 M-groups x 64 N-groups = 16384 waves.
// ---------------------------------------------------------------------------
__global__ void k_gemm_in(const __bf16* __restrict__ X,
                          const __bf16* __restrict__ WihB,
                          const float*  __restrict__ bias,
                          float* __restrict__ Gin) {
    int wave  = (blockIdx.x * (int)blockDim.x + (int)threadIdx.x) >> 5;
    int lane  = threadIdx.x & 31;
    int mg = wave >> 6;           // 0..255
    int ng = wave & 63;           // 0..63
    int m0 = mg * 32;
    int n0 = ng * 64;
    int lrow = lane & 15;
    int kg   = lane >> 4;

    v8f acc0 = {}, acc1 = {}, acc2 = {}, acc3 = {};
    v8f acc4 = {}, acc5 = {}, acc6 = {}, acc7 = {};

    const __bf16* pa = X    + (size_t)(m0 + lrow) * KDIM + kg * 8;
    const __bf16* pb = WihB + (size_t)(n0 + lrow) * KDIM + kg * 8;

    GEMM_K_LOOP(pa, pb, 16 * KDIM);

    float bv0 = bias[n0 +  0 + lrow];
    float bv1 = bias[n0 + 16 + lrow];
    float bv2 = bias[n0 + 32 + lrow];
    float bv3 = bias[n0 + 48 + lrow];

    // C/D layout: vgpr r, lane l -> (M = r + 8*(l>>4), N = l&15)
    #pragma unroll
    for (int r = 0; r < 8; ++r) {
        size_t base0 = (size_t)(m0 + r + 8 * kg) * G4H + n0 + lrow;
        Gin[base0 +  0] = acc0[r] + bv0;
        Gin[base0 + 16] = acc1[r] + bv1;
        Gin[base0 + 32] = acc2[r] + bv2;
        Gin[base0 + 48] = acc3[r] + bv3;
        size_t base1 = base0 + (size_t)16 * G4H;
        Gin[base1 +  0] = acc4[r] + bv0;
        Gin[base1 + 16] = acc5[r] + bv1;
        Gin[base1 + 32] = acc6[r] + bv2;
        Gin[base1 + 48] = acc7[r] + bv3;
    }
}

// ---------------------------------------------------------------------------
// Fused recurrent step: gates = Gin[t] + h @ Whh^T, then LSTM pointwise.
// Wave -> 2 batch tiles (32 rows) x one 16-col slice j of H x all FOUR gates
// (B reused across both M-tiles, A reused across 4 gates). Activation stays
// wave-local. 2 M-groups x 64 j-tiles = 128 waves.
// acc0..3 = gates i,f,g,o for M-tile 0; acc4..7 = gates for M-tile 1.
// ---------------------------------------------------------------------------
__global__ void k_lstm_step(const __bf16* __restrict__ hB,    // [64][1024] bf16
                            const __bf16* __restrict__ WhhB,  // [4096][1024] bf16
                            const float*  __restrict__ GinT,  // [64][4096] this step
                            float* __restrict__ cS,           // [64][1024] in/out
                            __bf16* __restrict__ hNext,       // [64][1024] bf16 out
                            float* __restrict__ hOut,         // d_out + t*B*H
                            float* __restrict__ hnOut,        // h_n (last step)
                            float* __restrict__ cnOut,        // c_n (last step)
                            int last) {
    int wave  = (blockIdx.x * (int)blockDim.x + (int)threadIdx.x) >> 5;
    int lane  = threadIdx.x & 31;
    int mg    = wave & 1;        // 0..1  (32-row batch group)
    int jtile = wave >> 1;       // 0..63 (hidden-column tile)
    int m0 = mg * 32;
    int j0 = jtile * 16;
    int lrow = lane & 15;
    int kg   = lane >> 4;
    int j    = j0 + lrow;

    // init accumulators from precomputed input-gate preactivations
    v8f acc0, acc1, acc2, acc3, acc4, acc5, acc6, acc7;
    #pragma unroll
    for (int r = 0; r < 8; ++r) {
        size_t base0 = (size_t)(m0 + r + 8 * kg) * G4H + j;
        acc0[r] = GinT[base0 + 0 * HID];
        acc1[r] = GinT[base0 + 1 * HID];
        acc2[r] = GinT[base0 + 2 * HID];
        acc3[r] = GinT[base0 + 3 * HID];
        size_t base1 = base0 + (size_t)16 * G4H;
        acc4[r] = GinT[base1 + 0 * HID];
        acc5[r] = GinT[base1 + 1 * HID];
        acc6[r] = GinT[base1 + 2 * HID];
        acc7[r] = GinT[base1 + 3 * HID];
    }

    const __bf16* pa = hB   + (size_t)(m0 + lrow) * KDIM + kg * 8;
    const __bf16* pb = WhhB + (size_t)j * KDIM + kg * 8;

    GEMM_K_LOOP(pa, pb, HID * KDIM);

    // pointwise LSTM update; each (lane, mt, r) owns a unique (b, j)
    #pragma unroll
    for (int r = 0; r < 8; ++r) {
        // M-tile 0
        {
            int b = m0 + r + 8 * kg;
            size_t idx = (size_t)b * HID + j;
            float iv = fsigmoid(acc0[r]);
            float fv = fsigmoid(acc1[r]);
            float gv = ftanh(acc2[r]);
            float ov = fsigmoid(acc3[r]);
            float cn = fv * cS[idx] + iv * gv;
            float hn = ov * ftanh(cn);
            cS[idx]    = cn;
            hOut[idx]  = hn;
            hNext[idx] = (__bf16)hn;
            if (last) { hnOut[idx] = hn; cnOut[idx] = cn; }
        }
        // M-tile 1
        {
            int b = m0 + 16 + r + 8 * kg;
            size_t idx = (size_t)b * HID + j;
            float iv = fsigmoid(acc4[r]);
            float fv = fsigmoid(acc5[r]);
            float gv = ftanh(acc6[r]);
            float ov = fsigmoid(acc7[r]);
            float cn = fv * cS[idx] + iv * gv;
            float hn = ov * ftanh(cn);
            cS[idx]    = cn;
            hOut[idx]  = hn;
            hNext[idx] = (__bf16)hn;
            if (last) { hnOut[idx] = hn; cnOut[idx] = cn; }
        }
    }
}

// ---------------------------------------------------------------------------
extern "C" void kernel_launch(void* const* d_in, const int* in_sizes, int n_in,
                              void* d_out, int out_size, void* d_ws, size_t ws_size,
                              hipStream_t stream) {
    const float* in_feat = (const float*)d_in[0];   // [64][512]
    const float* act     = (const float*)d_in[1];   // [128][64][512]
    const float* h0      = (const float*)d_in[2];   // [64][1024]
    const float* c0      = (const float*)d_in[3];   // [64][1024]
    const float* Wih     = (const float*)d_in[4];   // [4096][1024]
    const float* Whh     = (const float*)d_in[5];   // [4096][1024]
    const float* bih     = (const float*)d_in[6];   // [4096]
    const float* bhh     = (const float*)d_in[7];   // [4096]

    char* ws = (char*)d_ws;
    __bf16* WihB = (__bf16*)(ws + OFF_WIH);
    __bf16* WhhB = (__bf16*)(ws + OFF_WHH);
    __bf16* X    = (__bf16*)(ws + OFF_X);
    float*  Gin  = (float*)(ws + OFF_GIN);
    float*  bias = (float*)(ws + OFF_BIAS);
    __bf16* hB   = (__bf16*)(ws + OFF_H);
    float*  cS   = (float*)(ws + OFF_C);

    float* hs = (float*)d_out;                        // [128][64][1024]
    float* hn = hs + (size_t)L_STEPS * BATCH * HID;   // [64][1024]
    float* cn = hn + (size_t)BATCH * HID;             // [64][1024]

    // 1) weights -> bf16, fuse biases
    k_prep_weights<<<(G4H * KDIM) / 256, 256, 0, stream>>>(Wih, Whh, bih, bhh,
                                                           WihB, WhhB, bias);
    // 2) concat-inject X -> bf16
    k_prep_x<<<((size_t)MTOT * KDIM) / 256, 256, 0, stream>>>(in_feat, act, X);
    // 3) init recurrent state
    k_init_state<<<(BATCH * HID) / 256, 256, 0, stream>>>(h0, c0, hB, cS);
    // 4) hoisted input projection for all timesteps (WMMA), 16384 waves
    k_gemm_in<<<2048, 256, 0, stream>>>(X, WihB, bias, Gin);
    // 5) sequential fused recurrence (WMMA + activations), 128 waves/step
    for (int t = 0; t < L_STEPS; ++t) {
        const float* GinT = Gin + (size_t)t * BATCH * G4H;
        float* hOut = hs + (size_t)t * BATCH * HID;
        k_lstm_step<<<16, 256, 0, stream>>>(hB, WhhB, GinT, cS, hB, hOut,
                                            hn, cn, (t == L_STEPS - 1) ? 1 : 0);
    }
}
